// TCNPlus_60722247631741
// MI455X (gfx1250) — matrix operand order repaired and checked by reading, not compiled, not run on hardware
//
#include <hip/hip_runtime.h>
#include <hip/hip_bf16.h>

// ---------------------------------------------------------------------------
// Fused TCN for MI455X (gfx1250, wave32).
// One workgroup per (b,n) sample: z (128x256 f32) stays resident in LDS for
// all 8 layers. Pointwise GEMMs use v_wmma_f32_16x16x32_f16 (f16 in, f32 acc).
// Skip path evaluated only at t = T-1 (the only column the head consumes).
// ---------------------------------------------------------------------------

typedef __attribute__((ext_vector_type(16))) _Float16 v16h;
typedef __attribute__((ext_vector_type(8)))  float    v8f;

namespace {
constexpr int Bc   = 8;
constexpr int Tc   = 256;
constexpr int Nc   = 128;
constexpr int Dc   = 16;
constexpr int Hc   = 128;
constexpr int Hbc  = 64;
constexpr int Lc   = 8;
constexpr int OUTc = 24;

constexpr int ZLD = Tc + 4;   // padded row stride (floats) to avoid LDS bank conflicts
constexpr int ULD = Tc + 4;
constexpr int WLD = Tc + 4;

// LDS layout (bytes):
//  zs   : 128*ZLD f32      = 133120
//  us   : 64*ULD  f32      =  66560
//  wact : 64*WLD  f32      =  66560
//  w1h  : 64*128  f16      =  16384
//  w2h  : 128*64  f16      =  16384
//  dwl(384) gng(128) gnb(128) scol(128) skipl(128) y1(128) red(512) f32 = 6144
constexpr size_t SMEM_BYTES =
    (size_t)(128 * ZLD + 64 * ULD + 64 * WLD) * 4 + 2 * 8192 * 2 + 1536 * 4;
} // namespace

// --- WMMA fragment loaders (CDNA5 16-bit layouts, CDNA5 ISA) -----
// A 16x32 (MxK), f16 source in LDS (row stride ld):
//   lane L: m = L&15, half = L>>4 ; VGPR v holds K pair at
//   k = (v>>2)*16 + half*8 + (v&3)*2
__device__ __forceinline__ v16h load_a_f16(const _Float16* __restrict__ W,
                                           int ld, int m0, int k0, int lane) {
  v16h a;
  const int m  = m0 + (lane & 15);
  const int hf = lane >> 4;
#pragma unroll
  for (int v = 0; v < 8; ++v) {
    const int kk = k0 + ((v >> 2) << 4) + (hf << 3) + ((v & 3) << 1);
    a[2 * v]     = W[m * ld + kk];
    a[2 * v + 1] = W[m * ld + kk + 1];
  }
  return a;
}
// B 32x16 (KxN), f32 source in LDS (row stride ld), converted to f16:
//   lane L: n = L&15, half = L>>4 ; VGPR v holds k = (v>>2)*16 + half*8 + (v&3)*2 (+1)
__device__ __forceinline__ v16h load_b_f32(const float* __restrict__ M,
                                           int ld, int k0, int n0, int lane) {
  v16h b;
  const int n  = n0 + (lane & 15);
  const int hf = lane >> 4;
#pragma unroll
  for (int v = 0; v < 8; ++v) {
    const int k = k0 + ((v >> 2) << 4) + (hf << 3) + ((v & 3) << 1);
    b[2 * v]     = (_Float16)M[k * ld + n];
    b[2 * v + 1] = (_Float16)M[(k + 1) * ld + n];
  }
  return b;
}

__global__ __launch_bounds__(256)
void tcn_fused_kernel(const float* __restrict__ x,
                      const float* __restrict__ in_w,
                      const float* __restrict__ in_b,
                      const float* __restrict__ pw_in_w,
                      const float* __restrict__ dw_w,
                      const float* __restrict__ pw_out_w,
                      const float* __restrict__ gn_g,
                      const float* __restrict__ gn_b,
                      const float* __restrict__ skip_w,
                      const float* __restrict__ ln_g,
                      const float* __restrict__ ln_b,
                      const float* __restrict__ h1_w,
                      const float* __restrict__ h1_b,
                      const float* __restrict__ h2_w,
                      const float* __restrict__ h2_b,
                      float* __restrict__ out) {
  extern __shared__ char smem_raw[];
  float*    zs    = (float*)smem_raw;                  // 128 x ZLD
  float*    us    = zs + 128 * ZLD;                    // 64 x ULD
  float*    wact  = us + 64 * ULD;                     // 64 x WLD
  _Float16* w1h   = (_Float16*)(wact + 64 * WLD);      // 64 x 128
  _Float16* w2h   = w1h + 64 * 128;                    // 128 x 64
  float*    dwl   = (float*)(w2h + 128 * 64);          // 384
  float*    gng   = dwl + 384;                         // 128
  float*    gnb   = gng + 128;                         // 128
  float*    scol  = gnb + 128;                         // 128
  float*    skipl = scol + 128;                        // 128
  float*    y1    = skipl + 128;                       // 128
  float*    red   = y1 + 128;                          // 512

  const int tid  = threadIdx.x;
  const int lane = tid & 31;
  const int wave = tid >> 5;
  const int g    = blockIdx.x;       // 0 .. B*N-1
  const int bb   = g >> 7;
  const int nn   = g & 127;

  // ---- stage x slice (T x D) into LDS (reuse `us`), coalesced ----
  {
    const float* xbase = x + (size_t)bb * Tc * Nc * Dc + (size_t)nn * Dc;
    for (int idx = tid; idx < Tc * Dc; idx += 256) {
      const int t = idx >> 4, d = idx & 15;
      us[idx] = xbase[(size_t)t * Nc * Dc + d];
    }
  }
  if (tid < Hc) skipl[tid] = 0.f;
  __syncthreads();

  // ---- input projection: z[h][t] = in_b[h] + sum_d x[t][d] * in_w[d][h] ----
  for (int idx = tid; idx < Hc * Tc; idx += 256) {
    const int hh = idx >> 8, t = idx & 255;
    float acc = in_b[hh];
#pragma unroll
    for (int d = 0; d < Dc; ++d) acc += us[t * Dc + d] * in_w[d * Hc + hh];
    zs[hh * ZLD + t] = acc;
  }
  __syncthreads();

  // =========================== layer loop ===========================
  for (int li = 0; li < Lc; ++li) {
    const int dil = 1 << (li & 3);
    const float* W1g = pw_in_w  + (size_t)li * Hbc * Hc;
    const float* W2g = pw_out_w + (size_t)li * Hc * Hbc;

    // stage per-layer weights (f16 for WMMA A operands)
    for (int idx = tid; idx < Hbc * Hc; idx += 256) w1h[idx] = (_Float16)W1g[idx];
    for (int idx = tid; idx < Hc * Hbc; idx += 256) w2h[idx] = (_Float16)W2g[idx];
    for (int idx = tid; idx < 2 * Hbc * 3; idx += 256)
      dwl[idx] = dw_w[(size_t)li * 2 * Hbc * 3 + idx];
    if (tid < Hc) { gng[tid] = gn_g[li * Hc + tid]; gnb[tid] = gn_b[li * Hc + tid]; }
    __syncthreads();

    // ---- GEMM1: u(64x256) = W1(64x128) x z(128x256), K=128 ----
    for (int ntl = 0; ntl < 2; ++ntl) {
      const int nt = wave * 2 + ntl;
      v16h bfr[4];
#pragma unroll
      for (int kt = 0; kt < 4; ++kt)
        bfr[kt] = load_b_f32(zs, ZLD, kt * 32, nt * 16, lane);
#pragma unroll
      for (int mt = 0; mt < 4; ++mt) {
        v8f acc = {};
#pragma unroll
        for (int kt = 0; kt < 4; ++kt) {
          v16h af = load_a_f16(w1h, Hc, mt * 16, kt * 32, lane);
          acc = __builtin_amdgcn_wmma_f32_16x16x32_f16(
              false, af, false, bfr[kt], (short)0, acc, false, false);
        }
        const int n  = nt * 16 + (lane & 15);
        const int mb = mt * 16 + (lane >> 4) * 8;
#pragma unroll
        for (int r = 0; r < 8; ++r) us[(mb + r) * ULD + n] = acc[r];
      }
    }
    __syncthreads();

    // ---- dilated causal depthwise conv + tanh*sigmoid gate -> wact ----
    for (int idx = tid; idx < Hbc * Tc; idx += 256) {
      const int c = idx >> 8, t = idx & 255;
      const int uf = c >> 1;          // f-branch input channel
      const int ug = 32 + (c >> 1);   // gate-branch input channel
      float af = 0.f, ag = 0.f;
#pragma unroll
      for (int k = 0; k < 3; ++k) {
        const int tau = t - (2 - k) * dil;
        if (tau >= 0) {
          af += dwl[c * 3 + k]           * us[uf * ULD + tau];
          ag += dwl[(Hbc + c) * 3 + k]   * us[ug * ULD + tau];
        }
      }
      wact[c * WLD + t] = tanhf(af) * (1.f / (1.f + expf(-ag)));
    }
    __syncthreads();

    // ---- skip path, last column only: skipl += skip_w[li] @ (W2 @ w[:,T-1]) ----
    if (tid < Hc) {
      float a = 0.f;
      for (int c = 0; c < Hbc; ++c) a += W2g[tid * Hbc + c] * wact[c * WLD + (Tc - 1)];
      scol[tid] = a;
    }
    __syncthreads();
    if (tid < Hc) {
      const float* SW = skip_w + (size_t)li * Hc * Hc + (size_t)tid * Hc;
      float a = skipl[tid];
      for (int c = 0; c < Hc; ++c) a += SW[c] * scol[c];
      skipl[tid] = a;
    }

    // ---- GEMM2: z(128x256) += W2(128x64) x wact(64x256), K=64 ----
    for (int ntl = 0; ntl < 2; ++ntl) {
      const int nt = wave * 2 + ntl;
      v16h bfr[2];
#pragma unroll
      for (int kt = 0; kt < 2; ++kt)
        bfr[kt] = load_b_f32(wact, WLD, kt * 32, nt * 16, lane);
#pragma unroll
      for (int mt = 0; mt < 8; ++mt) {
        const int n  = nt * 16 + (lane & 15);
        const int mb = mt * 16 + (lane >> 4) * 8;
        v8f acc;
#pragma unroll
        for (int r = 0; r < 8; ++r) acc[r] = zs[(mb + r) * ZLD + n];
#pragma unroll
        for (int kt = 0; kt < 2; ++kt) {
          v16h af = load_a_f16(w2h, Hbc, mt * 16, kt * 32, lane);
          acc = __builtin_amdgcn_wmma_f32_16x16x32_f16(
              false, af, false, bfr[kt], (short)0, acc, false, false);
        }
#pragma unroll
        for (int r = 0; r < 8; ++r) zs[(mb + r) * ZLD + n] = acc[r];
      }
    }
    __syncthreads();

    // ---- group norm over all (H,T) of z, in place ----
    {
      float s = 0.f, s2 = 0.f;
      for (int idx = tid; idx < Hc * Tc; idx += 256) {
        const float v = zs[(idx >> 8) * ZLD + (idx & 255)];
        s += v; s2 += v * v;
      }
      red[tid] = s; red[256 + tid] = s2;
      __syncthreads();
      for (int off = 128; off > 0; off >>= 1) {
        if (tid < off) { red[tid] += red[tid + off]; red[256 + tid] += red[256 + tid + off]; }
        __syncthreads();
      }
      const float inv  = 1.f / (float)(Hc * Tc);
      const float mu   = red[0] * inv;
      const float var  = red[256] * inv - mu * mu;
      const float rstd = rsqrtf(var + 1e-5f);
      for (int idx = tid; idx < Hc * Tc; idx += 256) {
        const int hh = idx >> 8, t = idx & 255;
        zs[hh * ZLD + t] = (zs[hh * ZLD + t] - mu) * rstd * gng[hh] + gnb[hh];
      }
      __syncthreads();
    }
  }

  // =========================== head ===========================
  // LayerNorm over skipl[128]
  red[tid]       = (tid < Hc) ? skipl[tid]               : 0.f;
  red[256 + tid] = (tid < Hc) ? skipl[tid] * skipl[tid]  : 0.f;
  __syncthreads();
  for (int off = 128; off > 0; off >>= 1) {
    if (tid < off) { red[tid] += red[tid + off]; red[256 + tid] += red[256 + tid + off]; }
    __syncthreads();
  }
  {
    const float mu   = red[0] * (1.f / Hc);
    const float var  = red[256] * (1.f / Hc) - mu * mu;
    const float rstd = rsqrtf(var + 1e-5f);
    if (tid < Hc) scol[tid] = (skipl[tid] - mu) * rstd * ln_g[tid] + ln_b[tid];
  }
  __syncthreads();
  // h1 + exact GELU
  if (tid < Hc) {
    float a = h1_b[tid];
    for (int k = 0; k < Hc; ++k) a += scol[k] * h1_w[k * Hc + tid];
    y1[tid] = 0.5f * a * (1.f + erff(a * 0.70710678118654752440f));
  }
  __syncthreads();
  // h2 -> out[b][o][n]
  if (tid < OUTc) {
    float a = h2_b[tid];
    for (int j = 0; j < Hc; ++j) a += y1[j] * h2_w[j * OUTc + tid];
    out[((size_t)bb * OUTc + tid) * Nc + nn] = a;
  }
}

extern "C" void kernel_launch(void* const* d_in, const int* in_sizes, int n_in,
                              void* d_out, int out_size, void* d_ws, size_t ws_size,
                              hipStream_t stream) {
  (void)in_sizes; (void)n_in; (void)out_size; (void)d_ws; (void)ws_size;
  const float* x        = (const float*)d_in[0];
  const float* in_w     = (const float*)d_in[1];
  const float* in_b     = (const float*)d_in[2];
  const float* pw_in_w  = (const float*)d_in[3];
  const float* dw_w     = (const float*)d_in[4];
  const float* pw_out_w = (const float*)d_in[5];
  const float* gn_g     = (const float*)d_in[6];
  const float* gn_b     = (const float*)d_in[7];
  const float* skip_w   = (const float*)d_in[8];
  const float* ln_g     = (const float*)d_in[9];
  const float* ln_b     = (const float*)d_in[10];
  const float* h1_w     = (const float*)d_in[11];
  const float* h1_b     = (const float*)d_in[12];
  const float* h2_w     = (const float*)d_in[13];
  const float* h2_b     = (const float*)d_in[14];
  float* out = (float*)d_out;

  // Allow ~305KB dynamic LDS (CDNA5 WGP has 320KB); harmless if already set.
  (void)hipFuncSetAttribute((const void*)tcn_fused_kernel,
                            hipFuncAttributeMaxDynamicSharedMemorySize,
                            (int)SMEM_BYTES);

  tcn_fused_kernel<<<dim3(Bc * Nc), dim3(256), SMEM_BYTES, stream>>>(
      x, in_w, in_b, pw_in_w, dw_w, pw_out_w, gn_g, gn_b, skip_w,
      ln_g, ln_b, h1_w, h1_b, h2_w, h2_b, out);
}
